// LSTM_Cell_11398843204098
// MI455X (gfx1250) — matrix operand order repaired and checked by reading, not compiled, not run on hardware
//
#include <hip/hip_runtime.h>
#include <cstdint>
#include <cstddef>

typedef __attribute__((ext_vector_type(16))) __bf16 v16bf;
typedef __attribute__((ext_vector_type(8)))  float  v8f;

#define NB        131072          // batch
#define HID       128
#define KTOT      256             // IN + H
#define ROWS_PER_BLOCK 32
#define LDS_STRIDE 132            // uints per row: 128 K-pairs + 4 pad (bank-conflict-free)

union FragU {
    v16bf        v;
    unsigned int u[8];
};

struct WPtrs {
    const float* Wx[4];   // f, i, c, o  each [128,128]
    const float* Wh[4];
    const float* bx[4];
    const float* bh[4];
};

__device__ __forceinline__ float trunc_bf(float a) {
    unsigned int ua = __builtin_bit_cast(unsigned int, a) & 0xFFFF0000u;
    return __builtin_bit_cast(float, ua);
}
// pack bf16(a) (low 16) and bf16(b) (high 16) by truncation
__device__ __forceinline__ unsigned int pack_bf2(float a, float b) {
    unsigned int ua = __builtin_bit_cast(unsigned int, a);
    unsigned int ub = __builtin_bit_cast(unsigned int, b);
    return (ua >> 16) | (ub & 0xFFFF0000u);
}

// Branch-free HW transcendentals: v_exp_f32 + v_rcp_f32 (saturate via inf path)
#define LOG2E  1.4426950408889634f
__device__ __forceinline__ float fast_sigmoid(float x) {
    float e = __builtin_amdgcn_exp2f(-LOG2E * x);
    return __builtin_amdgcn_rcpf(1.0f + e);
}
__device__ __forceinline__ float fast_tanh(float x) {
    float e = __builtin_amdgcn_exp2f((2.0f * LOG2E) * x);
    return 1.0f - 2.0f * __builtin_amdgcn_rcpf(1.0f + e);
}

// ---------------------------------------------------------------------------
// Prep: pack W (concat [512,256], gate-major f,i,c,o) into WMMA B-operand
// fragments, split into bf16 hi/lo.  Fragment (tile t, kstep kk, lane):
//   n = lane&15 ; K block covered by lane = (lane>>4)*16 .. +15 (2 bf16/VGPR)
//   B[k][n] = W[t*16+n][kk*32 + k]
// Also fuses biases: bias[g] = bx[g] + bh[g].
// Launch: 32 blocks x 256 threads  (32 tiles * 8 ksteps * 32 lanes)
// ---------------------------------------------------------------------------
__global__ __launch_bounds__(256) void lstm_prep(WPtrs P,
                                                 unsigned int* __restrict__ Whi,
                                                 unsigned int* __restrict__ Wlo,
                                                 float* __restrict__ bias)
{
    int flat = blockIdx.x * 256 + threadIdx.x;
    int t    = flat >> 8;          // 0..31  N-tile (16 gate cols)
    int rem  = flat & 255;
    int kk   = rem >> 5;           // 0..7   K step (32 wide)
    int lane = rem & 31;

    int n    = lane & 15;
    int g    = t * 16 + n;         // gate column 0..511
    int gi   = g >> 7;             // which gate matrix (f,i,c,o)
    int row  = g & 127;            // row inside that matrix
    int kb   = kk * 32 + ((lane >> 4) << 4);   // absolute K base for this lane

    const float* wx = P.Wx[gi] + (size_t)row * 128;
    const float* wh = P.Wh[gi] + (size_t)row * 128;

    size_t base = ((size_t)(t * 8 + kk) * 32 + lane) * 8;
#pragma unroll
    for (int p = 0; p < 8; ++p) {
        int k0 = kb + 2 * p;
        int k1 = k0 + 1;
        float w0 = (k0 < 128) ? wx[k0] : wh[k0 - 128];
        float w1 = (k1 < 128) ? wx[k1] : wh[k1 - 128];
        float h0 = trunc_bf(w0), h1 = trunc_bf(w1);
        Whi[base + p] = pack_bf2(h0, h1);
        Wlo[base + p] = pack_bf2(w0 - h0, w1 - h1);
    }

    if (kk == 0 && lane < 16) {
        bias[g] = P.bx[gi][row] + P.bh[gi][row];
    }
}

// ---------------------------------------------------------------------------
// Main kernel: 256 threads = 8 waves, 32 batch rows per block.
// Wave w owns hidden columns [16w, 16w+16); its 4 N-tiles are gates f,i,c,o
// for those same columns -> epilogue is fully in-register.
// gates = act @ W^T via bf16x2 (3 products) WMMA, f32 accumulate.
// ---------------------------------------------------------------------------
__global__ __launch_bounds__(256) void lstm_main(const float* __restrict__ X,
                                                 const float* __restrict__ Hprev,
                                                 const float* __restrict__ Cprev,
                                                 const unsigned int* __restrict__ Whi,
                                                 const unsigned int* __restrict__ Wlo,
                                                 const float* __restrict__ bias,
                                                 float* __restrict__ outH,
                                                 float* __restrict__ outC)
{
    __shared__ unsigned int sAhi[ROWS_PER_BLOCK * LDS_STRIDE];
    __shared__ unsigned int sAlo[ROWS_PER_BLOCK * LDS_STRIDE];

    const int tid     = threadIdx.x;
    const int rowBase = blockIdx.x * ROWS_PER_BLOCK;

    // ---- Stage X||h rows into LDS, pre-split into packed bf16 hi/lo -------
    {
        int r = tid >> 3;                 // local row 0..31
        int q = tid & 7;
        const float4* xr = (const float4*)(X     + (size_t)(rowBase + r) * 128);
        const float4* hr = (const float4*)(Hprev + (size_t)(rowBase + r) * 128);
        unsigned int* dh = &sAhi[r * LDS_STRIDE];
        unsigned int* dl = &sAlo[r * LDS_STRIDE];
#pragma unroll
        for (int it = 0; it < 8; ++it) {
            int c4 = q + it * 8;                       // float4 index 0..63 (k = 4*c4)
            float4 v = (c4 < 32) ? xr[c4] : hr[c4 - 32];
            float hx = trunc_bf(v.x), hy = trunc_bf(v.y);
            float hz = trunc_bf(v.z), hw = trunc_bf(v.w);
            int kp = c4 * 2;                           // K-pair index
            *(uint2*)&dh[kp] = make_uint2(pack_bf2(hx, hy), pack_bf2(hz, hw));
            *(uint2*)&dl[kp] = make_uint2(pack_bf2(v.x - hx, v.y - hy),
                                          pack_bf2(v.z - hz, v.w - hw));
        }
    }
    __syncthreads();

    const int lane = tid & 31;
    const int wid  = tid >> 5;      // 0..7 : hidden cols [16*wid, 16*wid+16)
    const int lhi  = lane >> 4;     // half-wave
    const int ln   = lane & 15;

    v8f acc[2][4];
#pragma unroll
    for (int mt = 0; mt < 2; ++mt)
#pragma unroll
        for (int nt = 0; nt < 4; ++nt)
            acc[mt][nt] = (v8f)0.0f;

    // ---- K loop: 8 steps of 32 -------------------------------------------
#pragma unroll
    for (int kk = 0; kk < 8; ++kk) {
        // A fragments (16x32 bf16): lane holds M = ln, K at
        // kk*32 + lhi*16 (+0..15)  -> uints [kp0..kp0+7]
        FragU ah[2], al[2];
        int kp0 = kk * 16 + lhi * 8;
#pragma unroll
        for (int mt = 0; mt < 2; ++mt) {
            const unsigned int* sh = &sAhi[(mt * 16 + ln) * LDS_STRIDE + kp0];
            const unsigned int* sl = &sAlo[(mt * 16 + ln) * LDS_STRIDE + kp0];
            *(uint4*)&ah[mt].u[0] = *(const uint4*)&sh[0];
            *(uint4*)&ah[mt].u[4] = *(const uint4*)&sh[4];
            *(uint4*)&al[mt].u[0] = *(const uint4*)&sl[0];
            *(uint4*)&al[mt].u[4] = *(const uint4*)&sl[4];
        }
#pragma unroll
        for (int nt = 0; nt < 4; ++nt) {
            // tile t = nt*8 + wid  -> gate cols nt*128 + wid*16 + [0,16)
            size_t fb = ((size_t)((nt * 8 + wid) * 8 + kk) * 32 + lane) * 8;
            FragU bh, bl;
            *(uint4*)&bh.u[0] = *(const uint4*)(Whi + fb);
            *(uint4*)&bh.u[4] = *(const uint4*)(Whi + fb + 4);
            *(uint4*)&bl.u[0] = *(const uint4*)(Wlo + fb);
            *(uint4*)&bl.u[4] = *(const uint4*)(Wlo + fb + 4);
#pragma unroll
            for (int mt = 0; mt < 2; ++mt) {
                acc[mt][nt] = __builtin_amdgcn_wmma_f32_16x16x32_bf16(
                    false, ah[mt].v, false, bh.v, (short)0, acc[mt][nt], false, false);
                acc[mt][nt] = __builtin_amdgcn_wmma_f32_16x16x32_bf16(
                    false, ah[mt].v, false, bl.v, (short)0, acc[mt][nt], false, false);
                acc[mt][nt] = __builtin_amdgcn_wmma_f32_16x16x32_bf16(
                    false, al[mt].v, false, bh.v, (short)0, acc[mt][nt], false, false);
            }
        }
    }

    // ---- Epilogue: all 4 gates for (row, j) live in this lane ------------
    const int j = wid * 16 + ln;          // hidden column
    float bv[4];
#pragma unroll
    for (int nt = 0; nt < 4; ++nt) bv[nt] = bias[nt * 128 + j];

    // Issue all c_prev loads first so they clause & overlap the gate math.
    float cp[2][8];
#pragma unroll
    for (int mt = 0; mt < 2; ++mt)
#pragma unroll
        for (int r = 0; r < 8; ++r) {
            int row = rowBase + mt * 16 + lhi * 8 + r;
            cp[mt][r] = Cprev[(size_t)row * HID + j];
        }

#pragma unroll
    for (int mt = 0; mt < 2; ++mt) {
#pragma unroll
        for (int r = 0; r < 8; ++r) {
            int row = rowBase + mt * 16 + lhi * 8 + r;   // C/D layout: M = 8*lhi + r
            size_t off = (size_t)row * HID + j;
            float f  = acc[mt][0][r] + bv[0];
            float i  = acc[mt][1][r] + bv[1];
            float cg = acc[mt][2][r] + bv[2];
            float o  = acc[mt][3][r] + bv[3];
            float ft = fast_sigmoid(f);
            float it = fast_sigmoid(i);
            float ot = fast_sigmoid(o);
            float ct = ft * cp[mt][r] + it * fast_tanh(cg);
            outC[off] = ct;
            outH[off] = ot * fast_tanh(ct);
        }
    }
}

// ---------------------------------------------------------------------------
extern "C" void kernel_launch(void* const* d_in, const int* in_sizes, int n_in,
                              void* d_out, int out_size, void* d_ws, size_t ws_size,
                              hipStream_t stream) {
    (void)in_sizes; (void)n_in; (void)out_size; (void)ws_size;

    const float* X  = (const float*)d_in[0];
    const float* Hp = (const float*)d_in[1];
    const float* Cp = (const float*)d_in[2];

    // inputs 3..18: (W_xf,b_xf,W_hf,b_hf), (W_xi,...), (W_xc,...), (W_xo,...)
    WPtrs P;
    for (int gi = 0; gi < 4; ++gi) {
        int b = 3 + gi * 4;
        P.Wx[gi] = (const float*)d_in[b + 0];
        P.bx[gi] = (const float*)d_in[b + 1];
        P.Wh[gi] = (const float*)d_in[b + 2];
        P.bh[gi] = (const float*)d_in[b + 3];
    }

    // workspace: Whi (256KB) | Wlo (256KB) | bias (2KB)
    unsigned int* Whi  = (unsigned int*)d_ws;
    unsigned int* Wlo  = Whi + 32 * 8 * 32 * 8;
    float*        bias = (float*)(Wlo + 32 * 8 * 32 * 8);

    lstm_prep<<<32, 256, 0, stream>>>(P, Whi, Wlo, bias);

    float* outH = (float*)d_out;
    float* outC = outH + (size_t)NB * HID;
    lstm_main<<<NB / ROWS_PER_BLOCK, 256, 0, stream>>>(X, Hp, Cp, Whi, Wlo, bias,
                                                       outH, outC);
}
